// TernaryConv2D_5368709120114
// MI455X (gfx1250) — matrix operand order repaired and checked by reading, not compiled, not run on hardware
//
#include <hip/hip_runtime.h>

typedef __attribute__((ext_vector_type(16))) _Float16 v16h;
typedef __attribute__((ext_vector_type(8)))  _Float16 v8h;
typedef __attribute__((ext_vector_type(4)))  float    f32x4;
typedef __attribute__((ext_vector_type(8)))  float    v8f;

#define TTH   0.05f
#define HWD   112
#define CIN   128
#define COUT  256
#define NPIX  (32*112*112)        // 401408
#define NELEM ((size_t)NPIX*CIN)  // 51,380,224
#define LDA   40                  // LDS row stride in halfs (80B: 16B-aligned, bank-conflict-free)

// workspace layout (bytes); requires ws_size >= ~208 MB
#define WS_SCALE 0u                        // 256 f32
#define WS_ZPAD  1024u                     // 64B zero source for SAME-padding lanes
#define WS_WQ    4096u                     // 9*256*128 f16 ternary, [pos][oc][ic]
#define WS_XH    (2u*1024*1024)            // NELEM f16 = 102,760,448 B
#define WS_XL    (106u*1024*1024)          // NELEM f16

// ---- CDNA5 async global->LDS copy helpers (ASYNCcnt-tracked) ----
__device__ __forceinline__ unsigned lds_addr(const void* p) {
    return (unsigned)(unsigned long long)p;   // low 32 bits of flat LDS-aperture addr = LDS offset
}
__device__ __forceinline__ void async_cp32(unsigned lds, unsigned long long g) {
    // copies 32B: two b128 async copies; offset applies to BOTH lds and global sides
    asm volatile("global_load_async_to_lds_b128 %0, %1, off\n\t"
                 "global_load_async_to_lds_b128 %0, %1, off offset:16"
                 :: "v"(lds), "v"(g) : "memory");
}
__device__ __forceinline__ void wait_async6() { asm volatile("s_wait_asynccnt 6" ::: "memory"); }
__device__ __forceinline__ void wait_async0() { asm volatile("s_wait_asynccnt 0" ::: "memory"); }

// ---- prep: per-channel scale + zero pad source ----
__global__ __launch_bounds__(256)
void prep_scale(const float* __restrict__ w, float* __restrict__ scale,
                _Float16* __restrict__ zpad) {
    const int oc = threadIdx.x;
    float s = 0.f;
    for (int k = 0; k < 9 * CIN; ++k) s += fabsf(w[(size_t)k * COUT + oc]);
    scale[oc] = s * (1.0f / (9.0f * CIN));
    if (oc < 32) zpad[oc] = (_Float16)0.f;    // 64B of zeros
}

// ---- prep: ternarize weights into [pos][oc][ic] f16 ----
__global__ __launch_bounds__(256)
void prep_wq(const float* __restrict__ w, _Float16* __restrict__ wq) {
    const int idx = blockIdx.x * 256 + threadIdx.x;   // over 9*256*128
    const int pos = idx >> 15;
    const int rem = idx & 32767;
    const int oc  = rem >> 7;
    const int ic  = rem & 127;
    const float v = w[((size_t)(pos * CIN) + ic) * COUT + oc];
    wq[idx] = (fabsf(v) < TTH) ? (_Float16)0.f
            : ((v > 0.f) ? (_Float16)1.f : (_Float16)-1.f);
}

// ---- prep: x fp32 -> f16 hi/lo split (memory bound, done once) ----
__global__ __launch_bounds__(256)
void prep_x(const float* __restrict__ x, _Float16* __restrict__ xh,
            _Float16* __restrict__ xl) {
    const size_t i = ((size_t)blockIdx.x * 256 + threadIdx.x) * 8;
    const f32x4 a = __builtin_nontemporal_load((const f32x4*)(x + i));
    const f32x4 b = __builtin_nontemporal_load((const f32x4*)(x + i + 4));
    float v[8] = { a.x, a.y, a.z, a.w, b.x, b.y, b.z, b.w };
    v8h hi, lo;
    #pragma unroll
    for (int j = 0; j < 8; ++j) {
        const _Float16 h = (_Float16)v[j];
        hi[j] = h;
        lo[j] = (_Float16)(v[j] - (float)h);
    }
    *(v8h*)(xh + i) = hi;
    *(v8h*)(xl + i) = lo;
}

// ---- main: implicit-GEMM ternary conv, f16 WMMA, async double-buffered LDS ----
__global__ __launch_bounds__(256)
void tconv_wmma_kernel(const _Float16* __restrict__ xh, const _Float16* __restrict__ xl,
                       const _Float16* __restrict__ wq, const _Float16* __restrict__ zpad,
                       const float* __restrict__ scale, const float* __restrict__ bias,
                       float* __restrict__ out) {
    __shared__ _Float16 sAh[2][128 * LDA];
    __shared__ _Float16 sAl[2][128 * LDA];
    __shared__ _Float16 sB [2][128 * LDA];

    const int tid  = threadIdx.x;
    const int lane = tid & 31;
    const int wid  = tid >> 5;
    const int wm   = wid >> 2;             // wave M offset: 64*wm
    const int wn   = wid & 3;              // wave N offset: 32*wn

    const int m0  = blockIdx.x * 128;
    const int oc0 = blockIdx.y * 128;

    // copy-producer coords: thread -> (row p, 32B segment)
    const int p  = tid >> 1;               // 0..127 (pixel row for A, oc row for B)
    const int so = (tid & 1) << 5;         // byte sub-offset 0/32 within 64B row slice

    const int flat = m0 + p;
    const int pn   = flat / (HWD * HWD);
    const int prem = flat % (HWD * HWD);
    const int ph   = prem / HWD;
    const int pw   = prem % HWD;

    // per-buffer LDS byte addresses for this thread's copy destinations
    unsigned dAh[2], dAl[2], dB[2];
    #pragma unroll
    for (int b = 0; b < 2; ++b) {
        dAh[b] = lds_addr(&sAh[b][p * LDA]) + so;
        dAl[b] = lds_addr(&sAl[b][p * LDA]) + so;
        dB [b] = lds_addr(&sB [b][p * LDA]) + so;
    }
    const unsigned long long zsrc = (unsigned long long)(size_t)zpad + so;
    const unsigned long long wqrow =
        (unsigned long long)(size_t)(wq + (size_t)(oc0 + p) * CIN) + so;

    v8f acc[4][2];
    const v8f vzero = {};
    #pragma unroll
    for (int i = 0; i < 4; ++i)
        #pragma unroll
        for (int j = 0; j < 2; ++j) acc[i][j] = vzero;

    // issue copies for a given step into buffer `buf`
    auto issue = [&](int step, int buf) {
        const int pos = step >> 2;
        const int kh  = pos / 3;
        const int kw  = pos % 3;
        const int ic0 = (step & 3) << 5;                  // halfs
        const int hh  = ph + kh - 1;
        const int ww  = pw + kw - 1;
        unsigned long long gah, gal;
        if (hh >= 0 && hh < HWD && ww >= 0 && ww < HWD) {
            const size_t off = (((size_t)pn * HWD + hh) * HWD + ww) * CIN + ic0;
            gah = (unsigned long long)(size_t)(xh + off) + so;
            gal = (unsigned long long)(size_t)(xl + off) + so;
        } else {
            gah = zsrc; gal = zsrc;
        }
        const unsigned long long gb = wqrow + (size_t)(pos * (COUT * CIN) ) * 2 + ic0 * 2;
        async_cp32(dAh[buf], gah);
        async_cp32(dAl[buf], gal);
        async_cp32(dB [buf], gb);
    };

    issue(0, 0);                           // prologue

    for (int step = 0; step < 36; ++step) {
        const int cur = step & 1;
        if (step < 35) { issue(step + 1, cur ^ 1); wait_async6(); }
        else           { wait_async0(); }
        __syncthreads();                   // all waves' data for `cur` present

        const _Float16* Ah = sAh[cur];
        const _Float16* Al = sAl[cur];
        const _Float16* Bt = sB [cur];
        const int r16 = lane & 15;
        const int kg  = lane >> 4;

        v16h bf[2];
        #pragma unroll
        for (int nt = 0; nt < 2; ++nt) {   // B: lane -> oc row, k kg*8..+7 and 16+kg*8..+7 (matches A)
            const _Float16* q = &Bt[(wn * 32 + nt * 16 + r16) * LDA + kg * 8];
            *((f32x4*)&bf[nt])     = *((const f32x4*)q);
            *((f32x4*)&bf[nt] + 1) = *((const f32x4*)(q + 16));
        }
        #pragma unroll
        for (int mt = 0; mt < 4; ++mt) {
            const int row = wm * 64 + mt * 16 + r16;
            const _Float16* pah = &Ah[row * LDA + kg * 8];
            const _Float16* pal = &Al[row * LDA + kg * 8];
            v16h ah, al;
            *((f32x4*)&ah)     = *((const f32x4*)pah);        // K kb..kb+7
            *((f32x4*)&ah + 1) = *((const f32x4*)(pah + 16)); // K kb+16..kb+23
            *((f32x4*)&al)     = *((const f32x4*)pal);
            *((f32x4*)&al + 1) = *((const f32x4*)(pal + 16));
            #pragma unroll
            for (int nt = 0; nt < 2; ++nt) {
                acc[mt][nt] = __builtin_amdgcn_wmma_f32_16x16x32_f16(
                    false, ah, false, bf[nt], (short)0, acc[mt][nt], false, false);
                acc[mt][nt] = __builtin_amdgcn_wmma_f32_16x16x32_f16(
                    false, al, false, bf[nt], (short)0, acc[mt][nt], false, false);
            }
        }
        __syncthreads();                   // readers done before buffer reuse
    }

    // ---- epilogue: y = acc * scale[oc] + bias[oc], nontemporal stores ----
    const int ncol   = lane & 15;
    const int mshift = (lane >> 4) * 8;    // C/D layout: M = r + 8*(lane>=16)
    #pragma unroll
    for (int mt = 0; mt < 4; ++mt) {
        #pragma unroll
        for (int nt = 0; nt < 2; ++nt) {
            const int oc = oc0 + wn * 32 + nt * 16 + ncol;
            const float s  = scale[oc];
            const float bb = bias[oc];
            #pragma unroll
            for (int r = 0; r < 8; ++r) {
                const int row = m0 + wm * 64 + mt * 16 + mshift + r;
                __builtin_nontemporal_store(acc[mt][nt][r] * s + bb,
                                            &out[(size_t)row * COUT + oc]);
            }
        }
    }
}

extern "C" void kernel_launch(void* const* d_in, const int* in_sizes, int n_in,
                              void* d_out, int out_size, void* d_ws, size_t ws_size,
                              hipStream_t stream) {
    const float* x = (const float*)d_in[0];   // (32,112,112,128) fp32
    const float* w = (const float*)d_in[1];   // (3,3,128,256)    fp32
    const float* b = (const float*)d_in[2];   // (256,)           fp32
    float* out = (float*)d_out;

    char* ws = (char*)d_ws;                   // assumes ws_size >= ~208 MB
    float*    scale = (float*)   (ws + WS_SCALE);
    _Float16* zpad  = (_Float16*)(ws + WS_ZPAD);
    _Float16* wq    = (_Float16*)(ws + WS_WQ);
    _Float16* xh    = (_Float16*)(ws + WS_XH);
    _Float16* xl    = (_Float16*)(ws + WS_XL);

    prep_scale<<<dim3(1), dim3(256), 0, stream>>>(w, scale, zpad);
    prep_wq<<<dim3((9 * COUT * CIN) / 256), dim3(256), 0, stream>>>(w, wq);
    prep_x<<<dim3((unsigned)(NELEM / 2048)), dim3(256), 0, stream>>>(x, xh, xl);

    dim3 grid(NPIX / 128, COUT / 128, 1);
    tconv_wmma_kernel<<<grid, dim3(256), 0, stream>>>(xh, xl, wq, zpad, scale, b, out);
}
